// ScaledDotProductAttention_37890201485370
// MI455X (gfx1250) — compile-verified, matrix operand order checked by reading, not run on hardware
//
#include <hip/hip_runtime.h>
#include <hip/hip_bf16.h>

// Flash attention (causal) B=4,H=16,S=2048,D=128, fp32 I/O.
// bf16 WMMA (v_wmma_f32_16x16x32_bf16), gfx1250 wave32,
// double-buffered LDS staging, exp2-domain online softmax.

#define BHN     64      // B*H
#define S_LEN   2048
#define D_DIM   128
#define BM      64      // query rows per workgroup (16 per wave)
#define BN      32      // keys per inner block
#define NWAVES  4

typedef __attribute__((ext_vector_type(16))) __bf16 v16bf;
typedef __attribute__((ext_vector_type(8)))  __bf16 v8bf;
typedef __attribute__((ext_vector_type(4)))  __bf16 v4bf;
typedef __attribute__((ext_vector_type(8)))  float  v8f;
typedef __attribute__((ext_vector_type(4)))  float  v4f;

__device__ __forceinline__ v16bf cat16(v8bf lo, v8bf hi) {
  return __builtin_shufflevector(lo, hi, 0,1,2,3,4,5,6,7,8,9,10,11,12,13,14,15);
}

// Max-reduce across the 16-lane half-groups (lanes 0-15 / 16-31 independently).
__device__ __forceinline__ float redmax16(float v) {
  v = fmaxf(v, __int_as_float(__builtin_amdgcn_ds_swizzle(__float_as_int(v), 0x041f)));
  v = fmaxf(v, __int_as_float(__builtin_amdgcn_ds_swizzle(__float_as_int(v), 0x081f)));
  v = fmaxf(v, __int_as_float(__builtin_amdgcn_ds_swizzle(__float_as_int(v), 0x101f)));
  v = fmaxf(v, __int_as_float(__builtin_amdgcn_ds_swizzle(__float_as_int(v), 0x201f)));
  return v;
}
__device__ __forceinline__ float redsum16(float v) {
  v += __int_as_float(__builtin_amdgcn_ds_swizzle(__float_as_int(v), 0x041f));
  v += __int_as_float(__builtin_amdgcn_ds_swizzle(__float_as_int(v), 0x081f));
  v += __int_as_float(__builtin_amdgcn_ds_swizzle(__float_as_int(v), 0x101f));
  v += __int_as_float(__builtin_amdgcn_ds_swizzle(__float_as_int(v), 0x201f));
  return v;
}

// ---- K/V block staging helpers (prefetch to regs, convert+store to LDS) ----
// K: thread owns 1 row x 32 cols (row-major bf16 store).
// V: thread owns 4 rows x 8 cols, transposed in registers -> packed b64 stores.
__device__ __forceinline__ void load_kv_regs(const float* __restrict__ Kg,
                                             const float* __restrict__ Vg,
                                             int n0, int tid,
                                             v4f kreg[8], v4f vreg[8]) {
  const int kr = tid >> 2, kc = (tid & 3) * 32;
  const float* kp = Kg + (size_t)(n0 + kr) * D_DIM + kc;
#pragma unroll
  for (int j = 0; j < 8; ++j) kreg[j] = *(const v4f*)(kp + j * 4);
  const int vr = (tid >> 4) * 4, vc = (tid & 15) * 8;
  const float* vp = Vg + (size_t)(n0 + vr) * D_DIM + vc;
#pragma unroll
  for (int r = 0; r < 4; ++r) {
    vreg[r * 2 + 0] = *(const v4f*)(vp + (size_t)r * D_DIM);
    vreg[r * 2 + 1] = *(const v4f*)(vp + (size_t)r * D_DIM + 4);
  }
}

__device__ __forceinline__ void store_kv_lds(__bf16* ldsK, __bf16* ldsVt,
                                             int tid,
                                             const v4f kreg[8], const v4f vreg[8]) {
  const int kr = tid >> 2, kc = (tid & 3) * 32;
#pragma unroll
  for (int j = 0; j < 8; ++j) {
    v4bf t = { (__bf16)kreg[j][0], (__bf16)kreg[j][1],
               (__bf16)kreg[j][2], (__bf16)kreg[j][3] };
    *(v4bf*)&ldsK[kr * D_DIM + kc + j * 4] = t;
  }
  const int vr = (tid >> 4) * 4, vc = (tid & 15) * 8;
#pragma unroll
  for (int j = 0; j < 8; ++j) {
    const int h = j >> 2, e = j & 3;   // which v4f of the row / element in it
    v4bf t = { (__bf16)vreg[0 * 2 + h][e], (__bf16)vreg[1 * 2 + h][e],
               (__bf16)vreg[2 * 2 + h][e], (__bf16)vreg[3 * 2 + h][e] };
    *(v4bf*)&ldsVt[(vc + j) * BN + vr] = t;   // Vt[feature][key], 8B store
  }
}

__global__ __launch_bounds__(NWAVES * 32)
void fa_causal_bf16_wmma(const float* __restrict__ Q,
                         const float* __restrict__ K,
                         const float* __restrict__ V,
                         float* __restrict__ O) {
  __shared__ __align__(16) __bf16 ldsK[2][BN * D_DIM];     // 2 x 8 KB
  __shared__ __align__(16) __bf16 ldsVt[2][D_DIM * BN];    // 2 x 8 KB (Vt[d][key])
  __shared__ __align__(16) __bf16 ldsP[NWAVES * 16 * BN];  // 4 KB

  const int tid  = threadIdx.x;
  const int lane = tid & 31;
  const int wave = tid >> 5;
  const int ln   = lane & 15;
  const int lh   = lane >> 4;
  const int bh   = blockIdx.y;
  const int q0   = blockIdx.x * BM;

  const size_t base = (size_t)bh * S_LEN * D_DIM;
  const float* Qg = Q + base;
  const float* Kg = K + base;
  const float* Vg = V + base;
  float*       Og = O + base;

  // 1/sqrt(128) * log2(e): scores land directly in the exp2 domain.
  const float qscale = 0.12753139726f;

  // ---- Load Q into WMMA A-fragment layout, pre-scaled, bf16 ----
  const int qrow = q0 + wave * 16 + ln;
  v16bf qf[4];
#pragma unroll
  for (int c = 0; c < 4; ++c) {
    const int k0 = c * 32 + 8 * lh;
    const v4f* p0 = (const v4f*)(Qg + (size_t)qrow * D_DIM + k0);
    const v4f* p1 = (const v4f*)(Qg + (size_t)qrow * D_DIM + k0 + 16);
    v4f a0 = p0[0], a1 = p0[1];
    v4f b0 = p1[0], b1 = p1[1];
    v8bf lo = { (__bf16)(a0[0]*qscale), (__bf16)(a0[1]*qscale),
                (__bf16)(a0[2]*qscale), (__bf16)(a0[3]*qscale),
                (__bf16)(a1[0]*qscale), (__bf16)(a1[1]*qscale),
                (__bf16)(a1[2]*qscale), (__bf16)(a1[3]*qscale) };
    v8bf hi = { (__bf16)(b0[0]*qscale), (__bf16)(b0[1]*qscale),
                (__bf16)(b0[2]*qscale), (__bf16)(b0[3]*qscale),
                (__bf16)(b1[0]*qscale), (__bf16)(b1[1]*qscale),
                (__bf16)(b1[2]*qscale), (__bf16)(b1[3]*qscale) };
    qf[c] = cat16(lo, hi);
  }

  // ---- Online-softmax state (lrow kept as per-lane partial sums) ----
  const v8f vzero = {0.f, 0.f, 0.f, 0.f, 0.f, 0.f, 0.f, 0.f};
  v8f   acc[8];
  float mrow[8], lpart[8];
#pragma unroll
  for (int d = 0; d < 8; ++d) acc[d] = vzero;
#pragma unroll
  for (int r = 0; r < 8; ++r) { mrow[r] = -3.0e38f; lpart[r] = 0.f; }

  // ---- Prologue: stage block 0 ----
  v4f kreg[8], vreg[8];
  load_kv_regs(Kg, Vg, 0, tid, kreg, vreg);
  store_kv_lds(ldsK[0], ldsVt[0], tid, kreg, vreg);
  __syncthreads();

  const int nend = q0 + BM;                 // causal key limit for this block
  const int wmin = q0 + wave * 16;          // this wave's lowest query row
  int buf = 0;

  for (int n0 = 0; n0 < nend; n0 += BN, buf ^= 1) {
    const bool have_next = (n0 + BN) < nend;
    if (have_next) load_kv_regs(Kg, Vg, n0 + BN, tid, kreg, vreg);  // prefetch

    if (n0 <= wmin + 15) {   // block contributes to this wave's rows
      const __bf16* Kl = ldsK[buf];
      const __bf16* Vl = ldsVt[buf];

      // ---- Scores ----
      v8f sc[2];
#pragma unroll
      for (int nt = 0; nt < 2; ++nt) {
        v8f s = vzero;
#pragma unroll
        for (int c = 0; c < 4; ++c) {
          const int koff = (nt * 16 + ln) * D_DIM + c * 32 + 8 * lh;
          v8bf blo = *(const v8bf*)&Kl[koff];
          v8bf bhi = *(const v8bf*)&Kl[koff + 16];
          s = __builtin_amdgcn_wmma_f32_16x16x32_bf16(
                  false, qf[c], false, cat16(blo, bhi), (short)0, s, false, false);
        }
        sc[nt] = s;
      }

      // ---- Causal mask: only the (<=2) diagonal blocks need it ----
      if (n0 + BN - 1 > wmin) {
#pragma unroll
        for (int nt = 0; nt < 2; ++nt) {
          const int col = n0 + nt * 16 + ln;
#pragma unroll
          for (int r = 0; r < 8; ++r) {
            const int row = wmin + r + 8 * lh;
            if (col > row) sc[nt][r] = -3.0e38f;
          }
        }
      }

      // ---- Online softmax, exp2 domain ----
      float alpha[8];
#pragma unroll
      for (int r = 0; r < 8; ++r) {
        float mx   = redmax16(fmaxf(sc[0][r], sc[1][r]));
        float mnew = fmaxf(mrow[r], mx);
        alpha[r]   = exp2f(mrow[r] - mnew);
        float p0   = exp2f(sc[0][r] - mnew);
        float p1   = exp2f(sc[1][r] - mnew);
        sc[0][r] = p0; sc[1][r] = p1;
        lpart[r] = lpart[r] * alpha[r] + (p0 + p1);   // lane-partial sum
        mrow[r]  = mnew;
      }
#pragma unroll
      for (int d = 0; d < 8; ++d)
#pragma unroll
        for (int r = 0; r < 8; ++r) acc[d][r] *= alpha[r];

      // ---- P: C-layout -> LDS -> A-layout (same-wave DS ops are in-order) ----
      const int pw = wave * (16 * BN);
#pragma unroll
      for (int nt = 0; nt < 2; ++nt)
#pragma unroll
        for (int r = 0; r < 8; ++r)
          ldsP[pw + (r + 8 * lh) * BN + nt * 16 + ln] = (__bf16)sc[nt][r];

      v8bf plo = *(const v8bf*)&ldsP[pw + ln * BN + 8 * lh];
      v8bf phi = *(const v8bf*)&ldsP[pw + ln * BN + 8 * lh + 16];
      v16bf pa = cat16(plo, phi);

      // ---- O += P @ V ----
#pragma unroll
      for (int d = 0; d < 8; ++d) {
        const int voff = (d * 16 + ln) * BN + 8 * lh;
        v8bf vlo = *(const v8bf*)&Vl[voff];
        v8bf vhi = *(const v8bf*)&Vl[voff + 16];
        acc[d] = __builtin_amdgcn_wmma_f32_16x16x32_bf16(
                     false, pa, false, cat16(vlo, vhi), (short)0, acc[d], false, false);
      }
    }

    if (have_next) store_kv_lds(ldsK[buf ^ 1], ldsVt[buf ^ 1], tid, kreg, vreg);
    __syncthreads();   // one barrier per iteration (double-buffered)
  }

  // ---- Epilogue: finish l reduction, divide, store fp32 ----
  float inv[8];
#pragma unroll
  for (int r = 0; r < 8; ++r) inv[r] = __builtin_amdgcn_rcpf(redsum16(lpart[r]));
#pragma unroll
  for (int d = 0; d < 8; ++d) {
#pragma unroll
    for (int r = 0; r < 8; ++r) {
      const int row = wmin + r + 8 * lh;
      Og[(size_t)row * D_DIM + d * 16 + ln] = acc[d][r] * inv[r];
    }
  }
}

extern "C" void kernel_launch(void* const* d_in, const int* in_sizes, int n_in,
                              void* d_out, int out_size, void* d_ws, size_t ws_size,
                              hipStream_t stream) {
  (void)in_sizes; (void)n_in; (void)out_size; (void)d_ws; (void)ws_size;
  const float* q = (const float*)d_in[0];
  const float* k = (const float*)d_in[1];
  const float* v = (const float*)d_in[2];
  // d_in[3] is the causal mask; causality is computed analytically in-kernel.
  float* out = (float*)d_out;
  dim3 grid(S_LEN / BM, BHN, 1);
  dim3 block(NWAVES * 32, 1, 1);
  fa_causal_bf16_wmma<<<grid, block, 0, stream>>>(q, k, v, out);
}
